// SubjBasisGenerator_8486855377099
// MI455X (gfx1250) — matrix operand order repaired and checked by reading, not compiled, not run on hardware
//
#include <hip/hip_runtime.h>
#include <hip/hip_bf16.h>
#include <math.h>

// ---------------- constants ----------------
#define B_  4
#define NQ_ 416
#define NC_ 257
#define NCP_ 288   // NC padded to multiple of 32 (zero-filled)
#define D_  768
#define H_  6
#define DH_ 128
#define G_  104
#define R_  4
#define LORA_ 64
#define GL_ 6656   // G*LORA
#define EPSV 1e-5f

#define BM 128
#define BN 128
#define BK 32
#define BKP 40     // padded LDS K-stride (80B rows: conflict-free, 16B-aligned chunks)

typedef __attribute__((ext_vector_type(16))) __bf16 v16bf;
typedef __attribute__((ext_vector_type(8)))  float  v8f;
typedef int v4i_ __attribute__((vector_size(16)));

#if defined(__gfx1250__) && __has_builtin(__builtin_amdgcn_global_load_async_to_lds_b128)
#define HAVE_ASYNC 1
#else
#define HAVE_ASYNC 0
#endif

// 16-byte tile-staging copy: async-to-LDS on CDNA5, plain copy fallback
__device__ __forceinline__ void stage_chunk(const __bf16* g, __bf16* l) {
#if HAVE_ASYNC
  __builtin_amdgcn_global_load_async_to_lds_b128(
      (__attribute__((address_space(1))) v4i_*)g,
      (__attribute__((address_space(3))) v4i_*)l, 0, 0);
#else
  *(uint4*)l = *(const uint4*)g;
#endif
}
__device__ __forceinline__ void stage_fence() {
#if HAVE_ASYNC
#if __has_builtin(__builtin_amdgcn_s_wait_asynccnt)
  __builtin_amdgcn_s_wait_asynccnt(0);
#else
  asm volatile("s_wait_asynccnt 0" ::: "memory");
#endif
#endif
  __syncthreads();
}

// ---------------- WMMA fragment loaders (ISA 7.12.2 layouts) ----------------
// A 16x32 bf16: lanes 0-15 row M=lane hold K {0..7,16..23}; lanes 16-31 {8..15,24..31}
__device__ __forceinline__ v16bf frag_a(const __bf16* As, int row0) {
  int lane = threadIdx.x & 31;
  const __bf16* p = As + (row0 + (lane & 15)) * BKP + ((lane & 16) ? 8 : 0);
  v16bf a;
#pragma unroll
  for (int e = 0; e < 8; ++e) a[e] = p[e];          // one 16B run
#pragma unroll
  for (int e = 0; e < 8; ++e) a[8 + e] = p[16 + e]; // second 16B run
  return a;
}
// B stored K-contiguous: Bs[n][k]; lanes 0-15 col N=lane hold K {0..7,16..23}; lanes 16-31 {8..15,24..31}
__device__ __forceinline__ v16bf frag_b(const __bf16* Bs, int col0) {
  int lane = threadIdx.x & 31;
  const __bf16* p = Bs + (col0 + (lane & 15)) * BKP + ((lane & 16) ? 8 : 0);
  v16bf b;
#pragma unroll
  for (int e = 0; e < 8; ++e) b[e] = p[e];          // one 16B run
#pragma unroll
  for (int e = 0; e < 8; ++e) b[8 + e] = p[16 + e]; // second 16B run
  return b;
}

// ---------------- batched bf16 WMMA GEMM ----------------
// C[z] = alpha * A[z](MxK,row-major) * B[z]   where B stored transposed (N x K).
// K must be a multiple of 32 (operands zero-padded). Staging is unguarded:
// operand buffers carry slack rows. Batch offsets: (z/divH)*s1 + (z%divH)*s2.
__global__ __launch_bounds__(256)
void gemm_wmma_bf16(const __bf16* __restrict__ A, const __bf16* __restrict__ Bt,
                    float* __restrict__ C,
                    int M, int N, int K, int lda, int ldb, int ldc,
                    float alpha, int divH,
                    long long sA1, long long sA2,
                    long long sB1, long long sB2, long long sC) {
  __shared__ __bf16 As[BM * BKP];
  __shared__ __bf16 Bs[BN * BKP];

  int z = blockIdx.z;
  A  += (long long)(z / divH) * sA1 + (long long)(z % divH) * sA2;
  Bt += (long long)(z / divH) * sB1 + (long long)(z % divH) * sB2;
  C  += (long long)z * sC;

  int tid = threadIdx.x;
  int wave = tid >> 5;
  int wm = wave >> 1;       // 0..3 -> 32-row stripes
  int wn = wave & 1;        // 0..1 -> 64-col stripes
  int rowBase = blockIdx.y * BM;
  int colBase = blockIdx.x * BN;

  v8f acc[2][4] = {};

  for (int kt = 0; kt < K; kt += BK) {
    // stage A tile: 128 rows x 32 bf16 (4 x 16B chunks per row)
#pragma unroll 2
    for (int ch = tid; ch < (BM * BK) / 8; ch += 256) {
      int r = ch >> 2, c8 = (ch & 3) * 8;
      stage_chunk(A + (size_t)(rowBase + r) * lda + kt + c8, As + r * BKP + c8);
    }
    // stage B tile: 128 N-rows x 32 bf16 (already (N,K): contiguous rows)
#pragma unroll 2
    for (int ch = tid; ch < (BN * BK) / 8; ch += 256) {
      int r = ch >> 2, c8 = (ch & 3) * 8;
      stage_chunk(Bt + (size_t)(colBase + r) * ldb + kt + c8, Bs + r * BKP + c8);
    }
    stage_fence();

    v16bf bfr[4];
#pragma unroll
    for (int ni = 0; ni < 4; ++ni) bfr[ni] = frag_b(Bs, wn * 64 + ni * 16);
#pragma unroll
    for (int mi = 0; mi < 2; ++mi) {
      v16bf af = frag_a(As, wm * 32 + mi * 16);
#pragma unroll
      for (int ni = 0; ni < 4; ++ni) {
        acc[mi][ni] = __builtin_amdgcn_wmma_f32_16x16x32_bf16(
            false, af, false, bfr[ni], (short)0, acc[mi][ni], false, false);
      }
    }
    __syncthreads();
  }

  // store: C/D layout VGPR v -> M=v (lanes<16) / v+8 (lanes>=16); N=lane&15
  int lane = tid & 31;
  int nlo = lane & 15;
  int mo = (lane & 16) ? 8 : 0;
#pragma unroll
  for (int mi = 0; mi < 2; ++mi)
#pragma unroll
    for (int ni = 0; ni < 4; ++ni) {
      int cbase = colBase + wn * 64 + ni * 16 + nlo;
      if (cbase >= N) continue;
      int rbase = rowBase + wm * 32 + mi * 16 + mo;
#pragma unroll
      for (int v = 0; v < 8; ++v) {
        int rr = rbase + v;
        if (rr < M) C[(size_t)rr * ldc + cbase] = alpha * acc[mi][ni][v];
      }
    }
}

// ---------------- packing kernels ----------------
__global__ __launch_bounds__(256)
void pack_cast(const float* __restrict__ in, __bf16* __restrict__ out, long long n) {
  for (long long i = (long long)blockIdx.x * 256 + threadIdx.x; i < n;
       i += (long long)gridDim.x * 256)
    out[i] = (__bf16)in[i];
}
// out (N x K) = transpose of in (K x N)
__global__ __launch_bounds__(256)
void pack_T(const float* __restrict__ in, __bf16* __restrict__ out, int K, int N) {
  long long total = (long long)K * N;
  for (long long i = (long long)blockIdx.x * 256 + threadIdx.x; i < total;
       i += (long long)gridDim.x * 256) {
    int n = (int)(i / K), k = (int)(i % K);
    out[i] = (__bf16)in[(size_t)k * N + n];
  }
}
// ctxT[b, d, j(pad 288)] = ctx[b, j, d], zero-padded in j
__global__ __launch_bounds__(256)
void pack_ctxT(const float* __restrict__ ctx, __bf16* __restrict__ out) {
  long long total = (long long)B_ * D_ * NCP_;
  for (long long i = (long long)blockIdx.x * 256 + threadIdx.x; i < total;
       i += (long long)gridDim.x * 256) {
    int j = (int)(i % NCP_);
    int d = (int)((i / NCP_) % D_);
    int b = (int)(i / ((long long)NCP_ * D_));
    out[i] = (j < NC_) ? (__bf16)ctx[((size_t)b * NC_ + j) * D_ + d] : (__bf16)0.f;
  }
}

// ---------------- row LayerNorm (in-place) + optional bf16 copy ----------------
__global__ __launch_bounds__(256)
void ln_rows(float* __restrict__ X, const float* __restrict__ g,
             const float* __restrict__ b, int width, __bf16* __restrict__ outbf) {
  __shared__ float r1[256], r2[256];
  int tid = threadIdx.x;
  float* x = X + (size_t)blockIdx.x * width;
  float s = 0.f, s2 = 0.f;
  for (int i = tid; i < width; i += 256) { float v = x[i]; s += v; s2 += v * v; }
  r1[tid] = s; r2[tid] = s2;
  __syncthreads();
  for (int off = 128; off > 0; off >>= 1) {
    if (tid < off) { r1[tid] += r1[tid + off]; r2[tid] += r2[tid + off]; }
    __syncthreads();
  }
  float mean = r1[0] / width;
  float var = r2[0] / width - mean * mean;
  float rstd = rsqrtf(var + EPSV);
  for (int i = tid; i < width; i += 256) {
    float v = (x[i] - mean) * rstd * g[i] + b[i];
    x[i] = v;
    if (outbf) outbf[(size_t)blockIdx.x * width + i] = (__bf16)v;
  }
}

// ---------------- row softmax (in-place fp32) + padded bf16 copy ----------------
__global__ __launch_bounds__(256)
void softmax_rows(float* __restrict__ X, __bf16* __restrict__ Xbf,
                  int width, int ldo) {
  __shared__ float red[256];
  int tid = threadIdx.x;
  float* x = X + (size_t)blockIdx.x * width;
  float mx = -3.4e38f;
  for (int i = tid; i < width; i += 256) mx = fmaxf(mx, x[i]);
  red[tid] = mx; __syncthreads();
  for (int off = 128; off > 0; off >>= 1) {
    if (tid < off) red[tid] = fmaxf(red[tid], red[tid + off]);
    __syncthreads();
  }
  mx = red[0]; __syncthreads();
  float s = 0.f;
  for (int i = tid; i < width; i += 256) { float e = __expf(x[i] - mx); x[i] = e; s += e; }
  red[tid] = s; __syncthreads();
  for (int off = 128; off > 0; off >>= 1) {
    if (tid < off) red[tid] += red[tid + off];
    __syncthreads();
  }
  float inv = 1.f / red[0];
  for (int i = tid; i < ldo; i += 256) {
    float v = (i < width) ? x[i] * inv : 0.f;
    if (i < width) x[i] = v;
    Xbf[(size_t)blockIdx.x * ldo + i] = (__bf16)v;
  }
}

// ---------------- per-group Gram matrix Q_g = W_g W_g^T and col-sum ----------------
__global__ __launch_bounds__(256)
void conv_gram(const float* __restrict__ Wconv, float* __restrict__ Qg,
               float* __restrict__ wcm) {
  int g = blockIdx.x, tid = threadIdx.x;
  const float* W = Wconv + (size_t)g * LORA_ * D_;
  for (int l = tid; l < LORA_; l += 256) {
    float s = 0.f;
    for (int d = 0; d < D_; ++d) s += W[(size_t)l * D_ + d];
    wcm[g * LORA_ + l] = s;
  }
  for (int p = tid; p < LORA_ * LORA_; p += 256) {
    int l = p >> 6, l2 = p & 63;
    const float* wa = W + (size_t)l * D_;
    const float* wb = W + (size_t)l2 * D_;
    float s = 0.f;
    for (int d = 0; d < D_; ++d) s += wa[d] * wb[d];
    Qg[(size_t)g * LORA_ * LORA_ + p] = s;
  }
}

// ---------------- per-(b,g,j) LN stats of conv rows via quadratic form ----------------
__global__ __launch_bounds__(64)
void v_stats(const float* __restrict__ mid, const float* __restrict__ Qg,
             const float* __restrict__ wcm, float* __restrict__ meanb,
             float* __restrict__ rstdb) {
  int bg = blockIdx.x;               // B*G blocks
  int b = bg / G_, g = bg % G_;
  __shared__ float Qs[LORA_ * LORA_];
  __shared__ float ws[LORA_];
  int tid = threadIdx.x;
  for (int i = tid; i < LORA_ * LORA_; i += 64) Qs[i] = Qg[(size_t)g * LORA_ * LORA_ + i];
  if (tid < LORA_) ws[tid] = wcm[g * LORA_ + tid];
  __syncthreads();
  for (int j = tid; j < NC_; j += 64) {
    float m[LORA_];
    const float* mp = mid + (size_t)(b * NC_ + j) * GL_ + g * LORA_;
    float mean = 0.f;
#pragma unroll 8
    for (int l = 0; l < LORA_; ++l) { m[l] = mp[l]; mean += m[l] * ws[l]; }
    mean *= (1.f / (float)D_);
    float e2 = 0.f;
    for (int l = 0; l < LORA_; ++l) {
      float t = 0.f;
#pragma unroll 8
      for (int l2 = 0; l2 < LORA_; ++l2) t += Qs[l * LORA_ + l2] * m[l2];
      e2 += m[l] * t;
    }
    e2 *= (1.f / (float)D_);
    float var = e2 - mean * mean;
    int idx = (b * G_ + g) * NC_ + j;
    meanb[idx] = mean;
    rstdb[idx] = rsqrtf(var + EPSV);
  }
}

// ---------------- u[bh,r,g,:] = sum_j attn*rstd*mid ; s0,s1 scalars ----------------
__global__ __launch_bounds__(64)
void attn_reduce(const float* __restrict__ attn, const float* __restrict__ mid,
                 const float* __restrict__ meanb, const float* __restrict__ rstdb,
                 float* __restrict__ u, float* __restrict__ s0b,
                 float* __restrict__ s1b) {
  int blk = blockIdx.x;              // (B*H)*G blocks
  int bh = blk / G_, g = blk % G_;
  int b = bh / H_;
  int d = threadIdx.x;               // 0..63
  float uacc[R_] = {0.f, 0.f, 0.f, 0.f};
  float s0[R_] = {0.f, 0.f, 0.f, 0.f};
  float s1[R_] = {0.f, 0.f, 0.f, 0.f};
  for (int j = 0; j < NC_; ++j) {
    int sidx = (b * G_ + g) * NC_ + j;
    float rs = rstdb[sidx], mn = meanb[sidx];
    float mval = mid[(size_t)(b * NC_ + j) * GL_ + g * LORA_ + d];
#pragma unroll
    for (int r = 0; r < R_; ++r) {
      float a = attn[((size_t)bh * NQ_ + (r * G_ + g)) * NC_ + j];
      float aw = a * rs;
      uacc[r] += aw * mval;
      s0[r] += a;
      s1[r] += aw * mn;
    }
  }
#pragma unroll
  for (int r = 0; r < R_; ++r)
    u[((size_t)(bh * R_ + r) * G_ + g) * LORA_ + d] = uacc[r];
  if (d == 0) {
#pragma unroll
    for (int r = 0; r < R_; ++r) {
      int sidx = (bh * R_ + r) * G_ + g;
      s0b[sidx] = s0[r];
      s1b[sidx] = s1[r];
    }
  }
}

// ---------------- assemble merged-head output rows (bf16 for final GEMM) ----------------
__global__ __launch_bounds__(128)
void assemble_out(const float* __restrict__ u, const float* __restrict__ s0b,
                  const float* __restrict__ s1b, const float* __restrict__ cbuf,
                  const float* __restrict__ Wconv, const float* __restrict__ gv2,
                  const float* __restrict__ bv2, __bf16* __restrict__ outmbf) {
  int blk = blockIdx.x;              // (B*H)*NQ blocks
  int bh = blk / NQ_, i = blk % NQ_;
  int b = bh / H_, h = bh % H_;
  int g = i % G_, r = i / G_;
  int d = threadIdx.x;               // 0..127
  const float* uu = u + ((size_t)(bh * R_ + r) * G_ + g) * LORA_;
  const float* W = Wconv + (size_t)g * LORA_ * D_ + h * DH_ + d;
  float conv = 0.f;
#pragma unroll 8
  for (int l = 0; l < LORA_; ++l) conv += uu[l] * W[(size_t)l * D_];
  int sidx = (bh * R_ + r) * G_ + g;
  float s0 = s0b[sidx], s1 = s1b[sidx];
  int dd = h * DH_ + d;
  float val = gv2[dd] * (conv - s1) + bv2[dd] * s0 +
              cbuf[((size_t)bh * NQ_ + i) * DH_ + d];
  outmbf[((size_t)(b * NQ_ + i)) * D_ + dd] = (__bf16)val;
}

// ---------------- host launch ----------------
extern "C" void kernel_launch(void* const* d_in, const int* in_sizes, int n_in,
                              void* d_out, int out_size, void* d_ws, size_t ws_size,
                              hipStream_t stream) {
  const float* x     = (const float*)d_in[0];
  const float* ctx   = (const float*)d_in[1];
  const float* Wq    = (const float*)d_in[2];
  const float* gq    = (const float*)d_in[3];
  const float* bq    = (const float*)d_in[4];
  const float* Wk    = (const float*)d_in[5];
  const float* gk    = (const float*)d_in[6];
  const float* bk    = (const float*)d_in[7];
  const float* Wv    = (const float*)d_in[8];
  const float* gv1   = (const float*)d_in[9];
  const float* bv1   = (const float*)d_in[10];
  const float* Wconv = (const float*)d_in[11];
  const float* gv2   = (const float*)d_in[12];
  const float* bv2   = (const float*)d_in[13];
  const float* Wo    = (const float*)d_in[14];
  const float* go    = (const float*)d_in[15];
  const float* bo    = (const float*)d_in[16];
  float* out = (float*)d_out;
  (void)ws_size; (void)in_sizes; (void)n_in; (void)out_size;

  const int MQ = B_ * NQ_;   // 1664
  const int MC = B_ * NC_;   // 1028
  const int BH = B_ * H_;    // 24

  // fp32 workspace
  float* w = (float*)d_ws;
  size_t off = 0;
  auto alloc = [&](size_t n) { float* p = w + off; off += (n + 7) & ~(size_t)7; return p; };
  float* qb    = alloc((size_t)MQ * D_);
  float* kb    = alloc((size_t)MC * D_);
  float* midb  = alloc((size_t)MC * GL_);
  float* simb  = alloc((size_t)BH * NQ_ * NC_);
  float* cb    = alloc((size_t)BH * NQ_ * DH_);
  float* Qgb   = alloc((size_t)G_ * LORA_ * LORA_);
  float* wcmb  = alloc((size_t)G_ * LORA_);
  float* meanb = alloc((size_t)B_ * G_ * NC_);
  float* rstdb = alloc((size_t)B_ * G_ * NC_);
  float* ub    = alloc((size_t)BH * R_ * G_ * LORA_);
  float* s0b   = alloc((size_t)BH * R_ * G_);
  float* s1b   = alloc((size_t)BH * R_ * G_);

  // bf16 workspace (slack rows so unguarded tile staging stays in-bounds)
  __bf16* wb = (__bf16*)(w + off);
  size_t boff = 0;
  auto balloc = [&](size_t n) { __bf16* p = wb + boff; boff += (n + 15) & ~(size_t)15; return p; };
  __bf16* xbf    = balloc((size_t)MQ * D_);              // 1664 rows (13x128 exact)
  __bf16* cbf    = balloc((size_t)1152 * D_);            // ctx as A, 1028 -> 1152 rows
  __bf16* ctxT   = balloc((size_t)B_ * D_ * NCP_);       // (b, d, j pad 288)
  __bf16* WqT    = balloc((size_t)D_ * D_);
  __bf16* WkT    = balloc((size_t)D_ * D_);
  __bf16* WvT    = balloc((size_t)GL_ * D_);
  __bf16* WoT    = balloc((size_t)D_ * D_);
  __bf16* qbf    = balloc((size_t)(MQ + 96) * D_);       // per-(b,h) tiles read to row 511
  __bf16* kbf    = balloc((size_t)(MC + 132) * D_);      // B rows read to n=383 per batch
  __bf16* attnbf = balloc((size_t)(BH * NQ_ + 96) * NCP_);
  __bf16* outmbf = balloc((size_t)MQ * D_);

  dim3 blk(256);
  const float qkscale = 1.0f / sqrtf((float)DH_);   // (DH^-0.25)^2

  // ---- operand packing ----
  pack_cast<<<512, blk, 0, stream>>>(x, xbf, (long long)MQ * D_);
  pack_cast<<<512, blk, 0, stream>>>(ctx, cbf, (long long)MC * D_);
  pack_ctxT<<<512, blk, 0, stream>>>(ctx, ctxT);
  pack_T<<<512, blk, 0, stream>>>(Wq, WqT, D_, D_);
  pack_T<<<512, blk, 0, stream>>>(Wk, WkT, D_, D_);
  pack_T<<<1024, blk, 0, stream>>>(Wv, WvT, D_, GL_);
  pack_T<<<512, blk, 0, stream>>>(Wo, WoT, D_, D_);

  // ---- q = LN(x @ Wq) ----
  gemm_wmma_bf16<<<dim3(D_ / BN, MQ / BM, 1), blk, 0, stream>>>(
      xbf, WqT, qb, MQ, D_, D_, D_, D_, D_, 1.f, 1, 0, 0, 0, 0, 0);
  ln_rows<<<MQ, 256, 0, stream>>>(qb, gq, bq, D_, qbf);
  // ---- k = LN(ctx @ Wk) ----
  gemm_wmma_bf16<<<dim3(D_ / BN, (MC + BM - 1) / BM, 1), blk, 0, stream>>>(
      cbf, WkT, kb, MC, D_, D_, D_, D_, D_, 1.f, 1, 0, 0, 0, 0, 0);
  ln_rows<<<MC, 256, 0, stream>>>(kb, gk, bk, D_, kbf);
  // ---- mid = LN(ctx @ Wv) ----
  gemm_wmma_bf16<<<dim3(GL_ / BN, (MC + BM - 1) / BM, 1), blk, 0, stream>>>(
      cbf, WvT, midb, MC, GL_, D_, D_, D_, GL_, 1.f, 1, 0, 0, 0, 0, 0);
  ln_rows<<<MC, 256, 0, stream>>>(midb, gv1, bv1, GL_, (__bf16*)nullptr);
  // ---- Gram matrices + LN stats of conv rows ----
  conv_gram<<<G_, 256, 0, stream>>>(Wconv, Qgb, wcmb);
  v_stats<<<B_ * G_, 64, 0, stream>>>(midb, Qgb, wcmb, meanb, rstdb);
  // ---- sim[z=bh] = scale^2 * q_head @ k_head^T ----
  gemm_wmma_bf16<<<dim3((NC_ + BN - 1) / BN, (NQ_ + BM - 1) / BM, BH), blk, 0, stream>>>(
      qbf, kbf, simb, NQ_, NC_, DH_, D_, D_, NC_, qkscale, H_,
      (long long)NQ_ * D_, DH_, (long long)NC_ * D_, DH_, (long long)NQ_ * NC_);
  softmax_rows<<<BH * NQ_, 256, 0, stream>>>(simb, attnbf, NC_, NCP_);
  // ---- skip term: c[z=bh] = attn @ ctx_head  (K padded to 288, zero-filled) ----
  gemm_wmma_bf16<<<dim3(1, (NQ_ + BM - 1) / BM, BH), blk, 0, stream>>>(
      attnbf, ctxT, cb, NQ_, DH_, NCP_, NCP_, NCP_, DH_, 1.f, H_,
      (long long)H_ * NQ_ * NCP_, (long long)NQ_ * NCP_,
      (long long)D_ * NCP_, (long long)DH_ * NCP_, (long long)NQ_ * DH_);
  // ---- rank-64 attention-weighted reductions + output assembly ----
  attn_reduce<<<BH * G_, 64, 0, stream>>>(simb, midb, meanb, rstdb, ub, s0b, s1b);
  assemble_out<<<BH * NQ_, 128, 0, stream>>>(ub, s0b, s1b, cb, Wconv, gv2, bv2, outmbf);
  // ---- out = LN(outm @ Wo) ----
  gemm_wmma_bf16<<<dim3(D_ / BN, MQ / BM, 1), blk, 0, stream>>>(
      outmbf, WoT, out, MQ, D_, D_, D_, D_, D_, 1.f, 1, 0, 0, 0, 0, 0);
  ln_rows<<<MQ, 256, 0, stream>>>(out, go, bo, D_, (__bf16*)nullptr);
}
